// FaultGNN_24842090840586
// MI455X (gfx1250) — matrix operand order repaired and checked by reading, not compiled, not run on hardware
//
#include <hip/hip_runtime.h>
#include <hip/hip_bf16.h>

// ---------------- problem constants (match reference) ----------------
#define N_NODES  50000
#define N_EDGES  800000
#define G_GRAPHS 512
#define H_HEADS  4
#define C_CH     64
#define HID      256      // H*C
#define IN_F     128
#define OUT_F    8
#define NEG_SLOPE 0.2f
#define ATT_EPS   1e-16f

typedef __attribute__((ext_vector_type(16))) _Float16 v16h;
typedef __attribute__((ext_vector_type(8)))  float    v8f;

// Ordered-uint encoding so float max can use integer atomicMax.
__device__ __forceinline__ unsigned f2ord(float f) {
    unsigned u = __float_as_uint(f);
    return (u & 0x80000000u) ? ~u : (u | 0x80000000u);
}
__device__ __forceinline__ float ord2f(unsigned u) {
    return __uint_as_float((u & 0x80000000u) ? (u & 0x7fffffffu) : ~u);
}
#define ORD_NEG_INF 0x007fffffu   // f2ord(-inf)

__device__ __forceinline__ float lrelu(float x) { return (x > 0.f) ? x : NEG_SLOPE * x; }

// ---------------- utility kernels ----------------
__global__ void fill_f32(float* p, float v, int n) {
    int i = blockIdx.x * blockDim.x + threadIdx.x;
    if (i < n) p[i] = v;
}
__global__ void fill_u32(unsigned* p, unsigned v, int n) {
    int i = blockIdx.x * blockDim.x + threadIdx.x;
    if (i < n) p[i] = v;
}
__global__ void cvt_f32_to_f16(const float* __restrict__ in, _Float16* __restrict__ out, int n) {
    int i = blockIdx.x * blockDim.x + threadIdx.x;
    if (i < n) out[i] = (_Float16)in[i];
}

// ---------------- pack weight B[K x Ncols] (f32) into WMMA-B fragment layout (f16) ------------
// Fragment layout (16-bit B 32x16, contiguous k order, same as A): lane (g=lane>>4, n=lane&15) holds halves
// b[h] = B[k0 + 16g + h, col0 + n] for h = 0..15.  We store each lane's 16 halves
// contiguously: Bp[(((kb*tilesN) + ntile)*32 + lane)*16 + h], so the GEMM does one
// 32-byte vector load per fragment instead of 16 scalar loads.
__global__ void pack_b_f16(const float* __restrict__ W, _Float16* __restrict__ Bp,
                           int K, int Ncols) {
    int d = blockIdx.x * blockDim.x + threadIdx.x;
    if (d >= K * Ncols) return;
    int tilesN = Ncols >> 4;
    int h    = d & 15;
    int lane = (d >> 4) & 31;
    int rem  = d >> 9;
    int ntile = rem % tilesN;
    int kb    = rem / tilesN;
    int row = kb * 32 + (lane >> 4) * 16 + h;
    int col = ntile * 16 + (lane & 15);
    Bp[d] = (_Float16)W[(size_t)row * Ncols + col];
}

// ---------------- WMMA GEMM: C[M x Ncols] = A[M x K](f16) * B(packed f16) ----------------
// blockDim.x = (Ncols/64)*32 : each wave owns 4 consecutive N-tiles; one block covers a
// full 16-row x Ncols output stripe (blockIdx.y = M-tile). The 16x32 A-tile per k-step is
// staged once in LDS and shared by all waves; each wave issues 4 independent v_wmma per
// k-step (shared A fragment, 4 accumulators). Requires K%32==0, M%16==0, Ncols%64==0 —
// all shapes here comply, so EXEC is all-1s at every v_wmma (ISA 7.12 requirement).
__global__ void gemm_f16_wmma(const _Float16* __restrict__ A,
                              const _Float16* __restrict__ Bp,
                              float* __restrict__ C,
                              int K, int Ncols) {
    __shared__ unsigned ldsA[256];            // 16 rows x 32 halves
    const int lane = threadIdx.x & 31;
    const int wave = threadIdx.x >> 5;
    const int tilesN = Ncols >> 4;
    const int row0 = blockIdx.y * 16;
    const int m = lane & 15;
    const int g = lane >> 4;

    v8f acc0 = {}, acc1 = {}, acc2 = {}, acc3 = {};
    const unsigned* Au = (const unsigned*)A;
    const int nsteps = K >> 5;

    for (int kb = 0; kb < nsteps; ++kb) {
        const int k0 = kb * 32;
        // cooperative A-tile stage: 256 dwords = 16x32 halves
        for (int idx = threadIdx.x; idx < 256; idx += blockDim.x)
            ldsA[idx] = Au[((size_t)(row0 + (idx >> 4)) * K + k0) / 2 + (idx & 15)];
        __syncthreads();

        // A fragment (16-bit A 16x32, ISA 7.12.2): two contiguous 16B chunks per lane
        // -> compiler emits 2x ds_load_b128
        const _Float16* lA = (const _Float16*)ldsA;
        v16h a;
        #pragma unroll
        for (int r = 0; r < 8; ++r) {
            int up = 8 * g + r;
            a[2 * r]     = lA[m * 32 + 2 * up];
            a[2 * r + 1] = lA[m * 32 + 2 * up + 1];
        }
        // 4 packed B fragments: one 32B vector load each (2x global_load_b128)
        const v16h* bp = (const v16h*)Bp + (size_t)(kb * tilesN + wave * 4) * 32 + lane;
        v16h b0 = bp[0 * 32];
        v16h b1 = bp[1 * 32];
        v16h b2 = bp[2 * 32];
        v16h b3 = bp[3 * 32];
        acc0 = __builtin_amdgcn_wmma_f32_16x16x32_f16(false, a, false, b0, (short)0, acc0, false, false);
        acc1 = __builtin_amdgcn_wmma_f32_16x16x32_f16(false, a, false, b1, (short)0, acc1, false, false);
        acc2 = __builtin_amdgcn_wmma_f32_16x16x32_f16(false, a, false, b2, (short)0, acc2, false, false);
        acc3 = __builtin_amdgcn_wmma_f32_16x16x32_f16(false, a, false, b3, (short)0, acc3, false, false);
        __syncthreads();
    }
    // C/D layout: VGPR r -> row (r + 8g), col = lane&15 within tile
    const int colb = (wave * 4) * 16 + m;
    #pragma unroll
    for (int r = 0; r < 8; ++r) {
        float* crow = C + (size_t)(row0 + r + 8 * g) * Ncols + colb;
        crow[0]  = acc0[r];
        crow[16] = acc1[r];
        crow[32] = acc2[r];
        crow[48] = acc3[r];
    }
}

// ---------------- attention coefficients: s[n,h], d[n,h] ----------------
__global__ void attn_coef(const float* __restrict__ h, const float* __restrict__ as_,
                          const float* __restrict__ ad_, float* __restrict__ sv,
                          float* __restrict__ dv) {
    int i = blockIdx.x * blockDim.x + threadIdx.x;
    if (i >= N_NODES * H_HEADS) return;
    int nd = i >> 2, hh = i & 3;
    const float* hp = h + (size_t)nd * HID + hh * C_CH;
    const float* ap = as_ + hh * C_CH;
    const float* bp = ad_ + hh * C_CH;
    float s = 0.f, d = 0.f;
    #pragma unroll 8
    for (int c = 0; c < C_CH; ++c) { float v = hp[c]; s += v * ap[c]; d += v * bp[c]; }
    sv[i] = s; dv[i] = d;
}

// ---------------- edge pass A: segment max (all 4 heads per thread, float4) ----------------
__global__ void edge_max(const int* __restrict__ src, const int* __restrict__ dst,
                         const float* __restrict__ sv, const float* __restrict__ dv,
                         unsigned* __restrict__ mkey) {
    int e = blockIdx.x * blockDim.x + threadIdx.x;
    if (e >= N_EDGES) return;
    int s = src[e], t = dst[e];
    float4 a4 = *(const float4*)(sv + (size_t)s * 4);
    float4 b4 = *(const float4*)(dv + (size_t)t * 4);
    unsigned* mp = mkey + (size_t)t * 4;
    atomicMax(mp + 0, f2ord(lrelu(a4.x + b4.x)));
    atomicMax(mp + 1, f2ord(lrelu(a4.y + b4.y)));
    atomicMax(mp + 2, f2ord(lrelu(a4.z + b4.z)));
    atomicMax(mp + 3, f2ord(lrelu(a4.w + b4.w)));
}

// ---------------- edge pass B: exp + segment sum (all 4 heads per thread) ----------------
__global__ void edge_expsum(const int* __restrict__ src, const int* __restrict__ dst,
                            const float* __restrict__ sv, const float* __restrict__ dv,
                            const unsigned* __restrict__ mkey, float* __restrict__ denom,
                            float* __restrict__ wbuf) {
    int e = blockIdx.x * blockDim.x + threadIdx.x;
    if (e >= N_EDGES) return;
    int s = src[e], t = dst[e];
    float4 a4 = *(const float4*)(sv + (size_t)s * 4);
    float4 b4 = *(const float4*)(dv + (size_t)t * 4);
    const uint4 mk = *(const uint4*)(mkey + (size_t)t * 4);
    float4 w4;
    w4.x = __expf(lrelu(a4.x + b4.x) - ord2f(mk.x));
    w4.y = __expf(lrelu(a4.y + b4.y) - ord2f(mk.y));
    w4.z = __expf(lrelu(a4.z + b4.z) - ord2f(mk.z));
    w4.w = __expf(lrelu(a4.w + b4.w) - ord2f(mk.w));
    *(float4*)(wbuf + (size_t)e * 4) = w4;
    float* dp = denom + (size_t)t * 4;
    atomicAdd(dp + 0, w4.x);
    atomicAdd(dp + 1, w4.y);
    atomicAdd(dp + 2, w4.z);
    atomicAdd(dp + 3, w4.w);
}

// ---------------- edge pass C: weighted scatter (L2-resident atomics) ----------------
__global__ void edge_scatter(const int* __restrict__ src, const int* __restrict__ dst,
                             const float* __restrict__ wbuf, const float* __restrict__ denom,
                             const float* __restrict__ h, float* __restrict__ acc) {
    int i = blockIdx.x * blockDim.x + threadIdx.x;   // E*64 = 51.2M threads
    if (i >= N_EDGES * 64) return;
    int e = i >> 6, q = i & 63;
    int c0 = q * 4, hh = q >> 4;                      // head = c0/64
    int s = src[e], t = dst[e];
    const float* hp = h + (size_t)s * HID + c0;
    __builtin_prefetch(hp, 0, 0);                     // global_prefetch_b8
    float alpha = wbuf[(size_t)e * 4 + hh] / (denom[(size_t)t * 4 + hh] + ATT_EPS);
    float4 hv = *(const float4*)hp;
    float* op = acc + (size_t)t * HID + c0;
    atomicAdd(op + 0, hv.x * alpha);
    atomicAdd(op + 1, hv.y * alpha);
    atomicAdd(op + 2, hv.z * alpha);
    atomicAdd(op + 3, hv.w * alpha);
}

// ---------------- bias + BN(eval) + ReLU, emit f32 + f16 ----------------
__global__ void layer_fin(const float* __restrict__ acc, const float* __restrict__ bias,
                          const float* __restrict__ bns, const float* __restrict__ bnb,
                          const float* __restrict__ bnm, const float* __restrict__ bnv,
                          float* __restrict__ xf32, _Float16* __restrict__ xf16) {
    int i = blockIdx.x * blockDim.x + threadIdx.x;
    if (i >= N_NODES * HID) return;
    int c = i & 255;
    float v = acc[i] + bias[c];
    v = (v - bnm[c]) * bns[c] * rsqrtf(bnv[c] + 1e-5f) + bnb[c];
    v = fmaxf(v, 0.f);
    xf32[i] = v;
    xf16[i] = (_Float16)v;
}

// ---------------- global pooling ----------------
__global__ void pool_cnt(const int* __restrict__ batch, float* __restrict__ cnt) {
    int i = blockIdx.x * blockDim.x + threadIdx.x;
    if (i < N_NODES) atomicAdd(cnt + batch[i], 1.f);
}
__global__ void pool_acc(const int* __restrict__ batch, const float* __restrict__ x,
                         float* __restrict__ meanAcc, unsigned* __restrict__ maxKey) {
    int i = blockIdx.x * blockDim.x + threadIdx.x;
    if (i >= N_NODES * HID) return;
    int nd = i >> 8, c = i & 255;
    int b = batch[nd];
    float v = x[i];
    atomicAdd(meanAcc + (size_t)b * HID + c, v);
    atomicMax(maxKey + (size_t)b * HID + c, f2ord(v));
}
__global__ void pool_fin(const float* __restrict__ cnt, const float* __restrict__ meanAcc,
                         const unsigned* __restrict__ maxKey, float* __restrict__ g,
                         _Float16* __restrict__ g16) {
    int i = blockIdx.x * blockDim.x + threadIdx.x;
    if (i >= G_GRAPHS * HID) return;
    int gi = i >> 8, c = i & 255;
    float cn = cnt[gi];
    float mean = meanAcc[i] / fmaxf(cn, 1.f);
    float mx = (cn > 0.f) ? ord2f(maxKey[i]) : 0.f;
    int o1 = gi * 512 + c, o2 = o1 + 256;
    g[o1] = mean; g[o2] = mx;
    g16[o1] = (_Float16)mean; g16[o2] = (_Float16)mx;
}

// ---------------- MLP epilogue: bias + ReLU ----------------
__global__ void bias_relu(const float* __restrict__ acc, const float* __restrict__ bias,
                          int total, int cols, _Float16* out16, float* out32) {
    int i = blockIdx.x * blockDim.x + threadIdx.x;
    if (i >= total) return;
    float v = fmaxf(acc[i] + bias[i % cols], 0.f);
    if (out16) out16[i] = (_Float16)v;
    if (out32) out32[i] = v;
}

// ---------------- fused last linear (64->8) + log_softmax ----------------
__global__ void cls_final(const float* __restrict__ g2, const float* __restrict__ W,
                          const float* __restrict__ b, float* __restrict__ out) {
    int row = blockIdx.x * blockDim.x + threadIdx.x;
    if (row >= G_GRAPHS) return;
    float lg[OUT_F];
    #pragma unroll
    for (int o = 0; o < OUT_F; ++o) lg[o] = b[o];
    const float* gp = g2 + row * 64;
    for (int k = 0; k < 64; ++k) {
        float gv = gp[k];
        const float* wr = W + k * OUT_F;
        #pragma unroll
        for (int o = 0; o < OUT_F; ++o) lg[o] += gv * wr[o];
    }
    float mx = lg[0];
    #pragma unroll
    for (int o = 1; o < OUT_F; ++o) mx = fmaxf(mx, lg[o]);
    float se = 0.f;
    #pragma unroll
    for (int o = 0; o < OUT_F; ++o) se += __expf(lg[o] - mx);
    float lse = mx + __logf(se);
    #pragma unroll
    for (int o = 0; o < OUT_F; ++o) out[row * OUT_F + o] = lg[o] - lse;
}

// ---------------- host orchestration ----------------
static inline int ceil_div(int a, int b) { return (a + b - 1) / b; }

extern "C" void kernel_launch(void* const* d_in, const int* in_sizes, int n_in,
                              void* d_out, int out_size, void* d_ws, size_t ws_size,
                              hipStream_t stream) {
    (void)in_sizes; (void)n_in; (void)out_size; (void)ws_size;
    const float* x     = (const float*)d_in[0];
    const int*   ei    = (const int*)d_in[1];
    const int*   src   = ei;
    const int*   dst   = ei + N_EDGES;
    const int*   batch = (const int*)d_in[2];
    const float* W[3]   = {(const float*)d_in[3],  (const float*)d_in[11], (const float*)d_in[19]};
    const float* as_[3] = {(const float*)d_in[4],  (const float*)d_in[12], (const float*)d_in[20]};
    const float* ad_[3] = {(const float*)d_in[5],  (const float*)d_in[13], (const float*)d_in[21]};
    const float* bv[3]  = {(const float*)d_in[6],  (const float*)d_in[14], (const float*)d_in[22]};
    const float* bns[3] = {(const float*)d_in[7],  (const float*)d_in[15], (const float*)d_in[23]};
    const float* bnb[3] = {(const float*)d_in[8],  (const float*)d_in[16], (const float*)d_in[24]};
    const float* bnm[3] = {(const float*)d_in[9],  (const float*)d_in[17], (const float*)d_in[25]};
    const float* bnv[3] = {(const float*)d_in[10], (const float*)d_in[18], (const float*)d_in[26]};
    const float* cW0 = (const float*)d_in[27]; const float* cb0 = (const float*)d_in[28];
    const float* cW1 = (const float*)d_in[29]; const float* cb1 = (const float*)d_in[30];
    const float* cW2 = (const float*)d_in[31]; const float* cb2 = (const float*)d_in[32];

    // workspace carve-up (all offsets 256B aligned)
    char* ws = (char*)d_ws;
    size_t off = 0;
    auto carve = [&](size_t bytes) {
        void* p = ws + off;
        off += (bytes + 255) & ~size_t(255);
        return p;
    };
    _Float16* hf16  = (_Float16*)carve((size_t)N_NODES * HID * 2);   // layer input (f16)
    float*    hf32  = (float*)   carve((size_t)N_NODES * HID * 4);   // GEMM out / activations
    float*    accb  = (float*)   carve((size_t)N_NODES * HID * 4);   // message accumulator
    _Float16* wh16  = (_Float16*)carve((size_t)65536 * 2);           // packed f16 weight staging
    float*    sbuf  = (float*)   carve((size_t)N_NODES * H_HEADS * 4);
    float*    dbuf  = (float*)   carve((size_t)N_NODES * H_HEADS * 4);
    unsigned* mkey  = (unsigned*)carve((size_t)N_NODES * H_HEADS * 4);
    float*    denom = (float*)   carve((size_t)N_NODES * H_HEADS * 4);
    float*    wbuf  = (float*)   carve((size_t)N_EDGES * H_HEADS * 4);
    float*    cnt   = (float*)   carve((size_t)G_GRAPHS * 4);
    float*    mAcc  = (float*)   carve((size_t)G_GRAPHS * HID * 4);
    unsigned* xKey  = (unsigned*)carve((size_t)G_GRAPHS * HID * 4);
    float*    gfeat = (float*)   carve((size_t)G_GRAPHS * 512 * 4);
    _Float16* g16   = (_Float16*)carve((size_t)G_GRAPHS * 512 * 2);
    float*    macc1 = (float*)   carve((size_t)G_GRAPHS * 128 * 4);
    _Float16* g1_16 = (_Float16*)carve((size_t)G_GRAPHS * 128 * 2);
    float*    macc2 = (float*)   carve((size_t)G_GRAPHS * 64 * 4);
    float*    g2_32 = (float*)   carve((size_t)G_GRAPHS * 64 * 4);

    const int TB = 256;
    // x -> f16 (A operand stays row-major; staged through LDS in the GEMM)
    cvt_f32_to_f16<<<ceil_div(N_NODES * IN_F, TB), TB, 0, stream>>>(x, hf16, N_NODES * IN_F);

    for (int l = 0; l < 3; ++l) {
        const int K = (l == 0) ? IN_F : HID;
        // weights -> packed WMMA-B fragment layout (f16)
        pack_b_f16<<<ceil_div(K * HID, TB), TB, 0, stream>>>(W[l], wh16, K, HID);
        // h = X @ W via WMMA: block = (256/64)*32 = 128 thr, one 16xHID stripe per block
        gemm_f16_wmma<<<dim3(1, N_NODES / 16), (HID / 64) * 32, 0, stream>>>(hf16, wh16, hf32, K, HID);
        // attention coefficients
        attn_coef<<<ceil_div(N_NODES * H_HEADS, TB), TB, 0, stream>>>(hf32, as_[l], ad_[l], sbuf, dbuf);
        // init reductions
        fill_u32<<<ceil_div(N_NODES * H_HEADS, TB), TB, 0, stream>>>(mkey, ORD_NEG_INF, N_NODES * H_HEADS);
        fill_f32<<<ceil_div(N_NODES * H_HEADS, TB), TB, 0, stream>>>(denom, 0.f, N_NODES * H_HEADS);
        fill_f32<<<ceil_div(N_NODES * HID, TB), TB, 0, stream>>>(accb, 0.f, N_NODES * HID);
        // segment softmax + scatter
        edge_max    <<<ceil_div(N_EDGES, TB), TB, 0, stream>>>(src, dst, sbuf, dbuf, mkey);
        edge_expsum <<<ceil_div(N_EDGES, TB), TB, 0, stream>>>(src, dst, sbuf, dbuf, mkey, denom, wbuf);
        edge_scatter<<<ceil_div(N_EDGES * 64, TB), TB, 0, stream>>>(src, dst, wbuf, denom, hf32, accb);
        // bias + BN + ReLU -> next layer input (f32 for pooling, f16 for next GEMM)
        layer_fin<<<ceil_div(N_NODES * HID, TB), TB, 0, stream>>>(accb, bv[l], bns[l], bnb[l],
                                                                  bnm[l], bnv[l], hf32, hf16);
    }

    // pooling
    fill_f32<<<ceil_div(G_GRAPHS, TB), TB, 0, stream>>>(cnt, 0.f, G_GRAPHS);
    fill_f32<<<ceil_div(G_GRAPHS * HID, TB), TB, 0, stream>>>(mAcc, 0.f, G_GRAPHS * HID);
    fill_u32<<<ceil_div(G_GRAPHS * HID, TB), TB, 0, stream>>>(xKey, ORD_NEG_INF, G_GRAPHS * HID);
    pool_cnt<<<ceil_div(N_NODES, TB), TB, 0, stream>>>(batch, cnt);
    pool_acc<<<ceil_div(N_NODES * HID, TB), TB, 0, stream>>>(batch, hf32, mAcc, xKey);
    pool_fin<<<ceil_div(G_GRAPHS * HID, TB), TB, 0, stream>>>(cnt, mAcc, xKey, gfeat, g16);

    // classifier MLP (WMMA for 512x512x128 and 512x128x64)
    pack_b_f16<<<ceil_div(512 * 128, TB), TB, 0, stream>>>(cW0, wh16, 512, 128);
    gemm_f16_wmma<<<dim3(1, G_GRAPHS / 16), (128 / 64) * 32, 0, stream>>>(g16, wh16, macc1, 512, 128);
    bias_relu<<<ceil_div(G_GRAPHS * 128, TB), TB, 0, stream>>>(macc1, cb0, G_GRAPHS * 128, 128,
                                                              g1_16, (float*)nullptr);
    pack_b_f16<<<ceil_div(128 * 64, TB), TB, 0, stream>>>(cW1, wh16, 128, 64);
    gemm_f16_wmma<<<dim3(1, G_GRAPHS / 16), (64 / 64) * 32, 0, stream>>>(g1_16, wh16, macc2, 128, 64);
    bias_relu<<<ceil_div(G_GRAPHS * 64, TB), TB, 0, stream>>>(macc2, cb1, G_GRAPHS * 64, 64,
                                                             (_Float16*)nullptr, g2_32);
    cls_final<<<ceil_div(G_GRAPHS, TB), TB, 0, stream>>>(g2_32, cW2, cb2, (float*)d_out);
}
